// NTXentLoss_9036611191122
// MI455X (gfx1250) — matrix operand order repaired and checked by reading, not compiled, not run on hardware
//
#include <hip/hip_runtime.h>
#include <hip/hip_bf16.h>

#define B_ROWS  4096
#define C_DIM   512
#define N_ROWS  8192
#define INV_TAU 10.0f
#define EPSN    1e-12f

typedef _Float16 v8h  __attribute__((ext_vector_type(8)));
typedef _Float16 v16h __attribute__((ext_vector_type(16)));
typedef float    v8f  __attribute__((ext_vector_type(8)));

union FragAB { v16h v; v8h h[2]; };

__device__ __forceinline__ float lane_xor_add(float v, int mask) {
    int lane = threadIdx.x & 31;
    int o = __builtin_amdgcn_ds_bpermute(((lane ^ mask) << 2), __float_as_int(v));
    return v + __int_as_float(o);
}

// ---------------------------------------------------------------------------
// 1) Row L2-normalize fp32 -> f16 (z = [norm(z1); norm(z2)], row-major)
// ---------------------------------------------------------------------------
__global__ __launch_bounds__(256)
void ntx_normalize_kernel(const float* __restrict__ z1,
                          const float* __restrict__ z2,
                          _Float16* __restrict__ zh) {
    const int row = blockIdx.x;
    const float* src = (row < B_ROWS) ? (z1 + (size_t)row * C_DIM)
                                      : (z2 + (size_t)(row - B_ROWS) * C_DIM);
    const int t = threadIdx.x;
    float x0 = src[t];
    float x1 = src[t + 256];

    __shared__ float red[256];
    red[t] = x0 * x0 + x1 * x1;
    __syncthreads();
    for (int s = 128; s > 0; s >>= 1) {
        if (t < s) red[t] += red[t + s];
        __syncthreads();
    }
    float scale = 1.0f / fmaxf(sqrtf(red[0]), EPSN);
    _Float16* dst = zh + (size_t)row * C_DIM;
    dst[t]       = (_Float16)(x0 * scale);
    dst[t + 256] = (_Float16)(x1 * scale);
}

// ---------------------------------------------------------------------------
// 2) Zero the rowsum accumulator
// ---------------------------------------------------------------------------
__global__ void ntx_zero_kernel(float* __restrict__ p, int n) {
    int i = blockIdx.x * blockDim.x + threadIdx.x;
    if (i < n) p[i] = 0.0f;
}

// ---------------------------------------------------------------------------
// 3) Positives: pos[m] = exp(dot(z[m], z[(m+B)%N]) / tau)   (1 wave per row)
// ---------------------------------------------------------------------------
__global__ __launch_bounds__(256)
void ntx_pos_kernel(const _Float16* __restrict__ zh, float* __restrict__ pos) {
    const int row  = blockIdx.x * 8 + (threadIdx.x >> 5);
    const int lane = threadIdx.x & 31;
    const int prow = (row + B_ROWS) & (N_ROWS - 1);
    const _Float16* a = zh + (size_t)row  * C_DIM;
    const _Float16* b = zh + (size_t)prow * C_DIM;
    float acc = 0.0f;
    #pragma unroll
    for (int k = lane; k < C_DIM; k += 32)
        acc += (float)a[k] * (float)b[k];
    acc = lane_xor_add(acc, 16);
    acc = lane_xor_add(acc, 8);
    acc = lane_xor_add(acc, 4);
    acc = lane_xor_add(acc, 2);
    acc = lane_xor_add(acc, 1);
    if (lane == 0) pos[row] = __expf(acc * INV_TAU);
}

// ---------------------------------------------------------------------------
// 4) Fused symmetric Gram + exp + row-sum.
//    Block = 256 thr (8 waves); block tile 256(M) x 128(N).
//    Wave w owns rows [I*256 + w*32, +32) as 2 M-tiles x 8 N-tiles
//    (16 WMMA f32 accumulators, B fragment reused across both M-tiles).
//    Symmetry: only blocks with J >= 2I run. Per 128-row strip vs the
//    column strip: skip (below diag) / diag (row-sum only) /
//    above-diag (row-sum + transposed column-sum). S never hits memory.
// ---------------------------------------------------------------------------
__global__ __launch_bounds__(256)
void ntx_gram_rowsum_kernel(const _Float16* __restrict__ zh,
                            float* __restrict__ rowsum) {
    const int I = blockIdx.y;               // 256-row strip (0..31)
    const int J = blockIdx.x;               // 128-col strip (0..63)
    if (J < 2 * I) return;                  // strictly lower triangle: drop block

    const int wave  = threadIdx.x >> 5;
    const int lane  = threadIdx.x & 31;
    const int hi    = lane >> 4;            // lane half: 0 or 1
    const int l15   = lane & 15;

    const int rStrip = 2 * I + (wave >> 2); // this wave's 128-row strip index
    if (J < rStrip) return;                 // wave's tile below diagonal: skip
    const bool full = (J > rStrip);         // above diag: also emit col-sums

    const int mBase = I * 256 + wave * 32;  // wave covers 32 rows (2 M-tiles)
    const int nBase = J * 128;

    v8f acc[2][8] = {};                     // [mt][nt] 16x16 f32 tiles

    // A-frag per lane (ISA 16-bit A 16x32): row = mBase + mt*16 + l15;
    // halves[0..7] = K ka..ka+7, halves[8..15] = ka+16..ka+23, ka = k0 + 8*hi.
    const _Float16* aRow0 = zh + (size_t)(mBase + l15) * C_DIM + (hi ? 8 : 0);
    const _Float16* aRow1 = aRow0 + (size_t)16 * C_DIM;
    // B-frag per lane (32x16): B[k][n] = z[nBase+nt*16+n][k]; same K order as A:
    // halves[0..7] = K kb..kb+7, halves[8..15] = kb+16..kb+23, kb = k0 + 8*hi.
    const _Float16* bBase = zh + (size_t)(nBase + l15) * C_DIM + (hi ? 8 : 0);

    for (int k0 = 0; k0 < C_DIM; k0 += 32) {
        FragAB a0, a1;
        a0.h[0] = *(const v8h*)(aRow0 + k0);
        a0.h[1] = *(const v8h*)(aRow0 + k0 + 16);
        a1.h[0] = *(const v8h*)(aRow1 + k0);
        a1.h[1] = *(const v8h*)(aRow1 + k0 + 16);
        #pragma unroll
        for (int nt = 0; nt < 8; nt++) {
            const _Float16* bRow = bBase + (size_t)(nt * 16) * C_DIM + k0;
            FragAB b;
            b.h[0] = *(const v8h*)(bRow);
            b.h[1] = *(const v8h*)(bRow + 16);
            acc[0][nt] = __builtin_amdgcn_wmma_f32_16x16x32_f16(
                false, a0.v, false, b.v, (short)0, acc[0][nt], false, false);
            acc[1][nt] = __builtin_amdgcn_wmma_f32_16x16x32_f16(
                false, a1.v, false, b.v, (short)0, acc[1][nt], false, false);
        }
    }

    // Epilogue. C/D layout: VGPR r, lanes 0-15 -> (M=r, N=lane);
    // lanes 16-31 -> (M=r+8, N=lane-16). Every lane's elements share N=l15.
    float rs[2][8];
    float cs[8];
    #pragma unroll
    for (int nt = 0; nt < 8; nt++) cs[nt] = 0.0f;
    #pragma unroll
    for (int mt = 0; mt < 2; mt++)
        #pragma unroll
        for (int r = 0; r < 8; r++) rs[mt][r] = 0.0f;

    #pragma unroll
    for (int mt = 0; mt < 2; mt++)
        #pragma unroll
        for (int nt = 0; nt < 8; nt++)
            #pragma unroll
            for (int r = 0; r < 8; r++) {
                float e = __expf(acc[mt][nt][r] * INV_TAU);
                rs[mt][r] += e;   // partial row-sum (fixed m, this lane's n)
                cs[nt]    += e;   // partial col-sum (fixed n = l15)
            }

    // Direct row sums: reduce over the 16 lanes of each half.
    #pragma unroll
    for (int mt = 0; mt < 2; mt++)
        #pragma unroll
        for (int r = 0; r < 8; r++) {
            float t = rs[mt][r];
            t = lane_xor_add(t, 1);
            t = lane_xor_add(t, 2);
            t = lane_xor_add(t, 4);
            t = lane_xor_add(t, 8);
            if (l15 == 0)
                atomicAdd(rowsum + mBase + mt * 16 + r + hi * 8, t);
        }

    // Transposed contribution (S symmetric): column sums of this tile are
    // row-sum contributions for rows [nBase, nBase+128).
    if (full) {
        #pragma unroll
        for (int nt = 0; nt < 8; nt++) {
            float t = lane_xor_add(cs[nt], 16);  // combine m-halves
            if (hi == 0)
                atomicAdd(rowsum + nBase + nt * 16 + l15, t);
        }
    }
}

// ---------------------------------------------------------------------------
// 5) Final scalar: sum_r log((rowsum[r]-pos[r]) / pos[r]) / (N*B)
// ---------------------------------------------------------------------------
__global__ __launch_bounds__(256)
void ntx_loss_kernel(const float* __restrict__ rowsum,
                     const float* __restrict__ pos,
                     float* __restrict__ out) {
    __shared__ float red[256];
    const int t = threadIdx.x;
    float acc = 0.0f;
    for (int r = t; r < N_ROWS; r += 256) {
        float p = pos[r];
        acc += __logf((rowsum[r] - p) / p);   // == -log(p / neg)
    }
    red[t] = acc;
    __syncthreads();
    for (int s = 128; s > 0; s >>= 1) {
        if (t < s) red[t] += red[t + s];
        __syncthreads();
    }
    if (t == 0)
        out[0] = red[0] * (1.0f / ((float)N_ROWS * (float)B_ROWS));
}

// ---------------------------------------------------------------------------
extern "C" void kernel_launch(void* const* d_in, const int* in_sizes, int n_in,
                              void* d_out, int out_size, void* d_ws, size_t ws_size,
                              hipStream_t stream) {
    (void)in_sizes; (void)n_in; (void)out_size; (void)ws_size;
    const float* z1 = (const float*)d_in[0];
    const float* z2 = (const float*)d_in[1];
    float* out = (float*)d_out;

    char* ws = (char*)d_ws;
    _Float16* zh   = (_Float16*)ws;                                  // 8 MB
    float* rowsum  = (float*)(ws + (size_t)N_ROWS * C_DIM * 2);      // 32 KB
    float* pos     = rowsum + N_ROWS;                                // 32 KB

    ntx_normalize_kernel<<<N_ROWS, 256, 0, stream>>>(z1, z2, zh);
    ntx_zero_kernel<<<(N_ROWS + 255) / 256, 256, 0, stream>>>(rowsum, N_ROWS);
    ntx_pos_kernel<<<N_ROWS / 8, 256, 0, stream>>>(zh, pos);

    dim3 grid(N_ROWS / 128, N_ROWS / 256);  // 64 col strips x 32 row strips
    ntx_gram_rowsum_kernel<<<grid, 256, 0, stream>>>(zh, rowsum);

    ntx_loss_kernel<<<1, 256, 0, stream>>>(rowsum, pos, out);
}
